// WeaklySelector_84928683311766
// MI455X (gfx1250) — matrix operand order repaired and checked by reading, not compiled, not run on hardware
//
#include <hip/hip_runtime.h>
#include <hip/hip_bf16.h>
#include <stdint.h>

// ---------------------------------------------------------------------------
// Problem constants
// ---------------------------------------------------------------------------
#define B_    32
#define S_    4096
#define C_    512
#define K_    200
#define KP    208            // K padded to 13 tiles of 16
#define NSEL  256
#define NKT   13             // k tiles  (208/16)
#define NCT   16             // c tiles  (512/32)
#define NROWS (B_*S_)        // 131072 rows of x

typedef __attribute__((ext_vector_type(16))) __bf16 v16bf;
typedef __attribute__((ext_vector_type(8)))  float  v8f;

// ---------------------------------------------------------------------------
// Workspace layout (bytes)
// ---------------------------------------------------------------------------
#define OFF_LOGITS 0ull
#define SZ_LOGITS  ((size_t)NROWS * KP * 4)                 // 109,051,904
#define OFF_WFRAG  (OFF_LOGITS + SZ_LOGITS)
#define SZ_WFRAG   ((size_t)NKT * NCT * 32 * 32 * 2)        // 425,984
#define OFF_COLSUM (OFF_WFRAG + SZ_WFRAG)
#define SZ_COLSUM  ((size_t)B_ * 256 * 4)
#define OFF_MAXID  (OFF_COLSUM + SZ_COLSUM)
#define SZ_MAXID   1024ull
#define OFF_P      (OFF_MAXID + SZ_MAXID)
#define SZ_P       ((size_t)NROWS * 4)
#define OFF_RANKS  (OFF_P + SZ_P)

// ---------------------------------------------------------------------------
// 1) Pre-swizzle W^T into bf16 hi/lo A-fragments, laid out in a contiguous-k
//    (k = 16h + s) 16x32 A-fragment order, so the GEMM hot loop reads them as two
//    aligned v16bf loads per (ktile,ctile). One-time, L2-resident (0.43 MB).
//    A[m=kq][k=c]: lane holds M=lane%16; element e -> c_local =
//      e + (lane>=16 ? 16 : 0)
// ---------------------------------------------------------------------------
__global__ void prep_wfrag(const float* __restrict__ W, __bf16* __restrict__ wfrag)
{
    int kt = blockIdx.x, ct = blockIdx.y, lane = threadIdx.x;   // 32 threads
    int kq = lane & 15;
    int k  = kt * 16 + kq;
    __bf16* dst = wfrag + ((size_t)(kt * NCT + ct) * 32 + lane) * 32;
#pragma unroll
    for (int e = 0; e < 16; ++e) {
        int cl = e + ((lane >= 16) ? 16 : 0);
        int c  = ct * 32 + cl;
        float w = (k < K_) ? W[(size_t)c * K_ + k] : 0.0f;
        __bf16 h = (__bf16)w;
        __bf16 l = (__bf16)(w - (float)h);
        dst[e]      = h;
        dst[16 + e] = l;
    }
}

__global__ void zero_colsum(float* __restrict__ colsum)
{
    colsum[blockIdx.x * 256 + threadIdx.x] = 0.0f;
}

// ---------------------------------------------------------------------------
// 2) GEMM: logits^T tile = W^T (A) x x^T (B), bf16x3 split, fp32 accumulate.
//    One wave per 16 x-rows; accumulates all 13 k-tiles (104 acc VGPRs).
//    B-fragment (32x16, 16-bit): lane = 16*(c>=16 half) + row%16 holds 16
//    CONSECUTIVE c-values of one x row -> 4x b128 vector loads.
//    Epilogue: D element r at lane l is (k = kt*16 + r + 8*(l>=16),
//    row = rowbase + l%16); add bias, store two float4s per k-tile.
// ---------------------------------------------------------------------------
__global__ void __launch_bounds__(256)
gemm_wmma(const float* __restrict__ x, const float* __restrict__ bias,
          const __bf16* __restrict__ wfrag, float* __restrict__ logits)
{
    const int lane = threadIdx.x & 31;
    const int wave = threadIdx.x >> 5;
    const int rowtile = blockIdx.x * 8 + wave;
    const int myrow   = rowtile * 16 + (lane & 15);
    const float* xrow = x + (size_t)myrow * C_ + ((lane >= 16) ? 16 : 0);

    const v8f zero8 = {0.f, 0.f, 0.f, 0.f, 0.f, 0.f, 0.f, 0.f};
    v8f acc[NKT];
#pragma unroll
    for (int kt = 0; kt < NKT; ++kt) acc[kt] = zero8;

    for (int ct = 0; ct < NCT; ++ct) {
        const float4* xp = (const float4*)(xrow + ct * 32);
        float4 xa = xp[0], xb = xp[1], xc = xp[2], xd = xp[3];
        float xv[16] = {xa.x, xa.y, xa.z, xa.w, xb.x, xb.y, xb.z, xb.w,
                        xc.x, xc.y, xc.z, xc.w, xd.x, xd.y, xd.z, xd.w};
        v16bf bhi, blo;
#pragma unroll
        for (int e = 0; e < 16; ++e) {
            __bf16 h = (__bf16)xv[e];
            bhi[e] = h;
            blo[e] = (__bf16)(xv[e] - (float)h);
        }
        const __bf16* wbase = wfrag + ((size_t)ct * 32 + lane) * 32;
#pragma unroll
        for (int kt = 0; kt < NKT; ++kt) {
            const v16bf* wp = (const v16bf*)(wbase + (size_t)kt * NCT * 32 * 32);
            v16bf ahi = wp[0];
            v16bf alo = wp[1];
            acc[kt] = __builtin_amdgcn_wmma_f32_16x16x32_bf16(
                false, ahi, false, bhi, (short)0, acc[kt], false, false);
            acc[kt] = __builtin_amdgcn_wmma_f32_16x16x32_bf16(
                false, ahi, false, blo, (short)0, acc[kt], false, false);
            acc[kt] = __builtin_amdgcn_wmma_f32_16x16x32_bf16(
                false, alo, false, bhi, (short)0, acc[kt], false, false);
        }
    }

    const int koff = (lane >= 16) ? 8 : 0;
    float* lrow = logits + (size_t)myrow * KP;
#pragma unroll
    for (int kt = 0; kt < NKT; ++kt) {
        int k0 = kt * 16 + koff;
        if (k0 >= K_) continue;            // only kt==12, upper half (k=200..207)
        float4 lo4 = {acc[kt][0] + bias[k0 + 0], acc[kt][1] + bias[k0 + 1],
                      acc[kt][2] + bias[k0 + 2], acc[kt][3] + bias[k0 + 3]};
        float4 hi4 = {acc[kt][4] + bias[k0 + 4], acc[kt][5] + bias[k0 + 5],
                      acc[kt][6] + bias[k0 + 6], acc[kt][7] + bias[k0 + 7]};
        *(float4*)(lrow + k0)     = lo4;
        *(float4*)(lrow + k0 + 4) = hi4;
    }
}

// ---------------------------------------------------------------------------
// 3) Column sums over s per batch (argmax(mean) == argmax(sum)).
// ---------------------------------------------------------------------------
__global__ void colsum_kernel(const float* __restrict__ logits, float* __restrict__ colsum)
{
    int b = blockIdx.x, chunk = blockIdx.y, k = threadIdx.x;
    if (k >= K_) return;
    const int CH = S_ / 16;                 // 256 s-rows per chunk
    const float* base = logits + ((size_t)(b * S_ + chunk * CH)) * KP + k;
    float local = 0.f;
    for (int i = 0; i < CH; ++i) local += base[(size_t)i * KP];
    atomicAdd(&colsum[b * 256 + k], local);
}

// 4) Per-batch argmax with first-index tie-break (matches jnp.argmax).
__global__ void argmax_kernel(const float* __restrict__ colsum, int* __restrict__ maxid)
{
    int b = blockIdx.x, lane = threadIdx.x;      // 32 threads = 1 wave
    float bestv = -INFINITY; int besti = 0x7fffffff;
    for (int k = lane; k < K_; k += 32) {
        float v = colsum[b * 256 + k];
        if (v > bestv) { bestv = v; besti = k; }
    }
    for (int off = 16; off > 0; off >>= 1) {
        float ov = __shfl_down(bestv, off);
        int   oi = __shfl_down(besti, off);
        if (ov > bestv || (ov == bestv && oi < besti)) { bestv = ov; besti = oi; }
    }
    if (lane == 0) maxid[b] = besti;
}

// 5) One wave per row: softmax prob at the probe column.
__global__ void row_softmax(const float* __restrict__ logits,
                            const int* __restrict__ maxid, float* __restrict__ p)
{
    int lane = threadIdx.x & 31, wave = threadIdx.x >> 5;
    int row = blockIdx.x * 8 + wave;
    int b = row >> 12;
    const float* lrow = logits + (size_t)row * KP;
    float m = -INFINITY;
    for (int k = lane; k < K_; k += 32) m = fmaxf(m, lrow[k]);
    for (int off = 16; off > 0; off >>= 1) m = fmaxf(m, __shfl_xor(m, off));
    float z = 0.f;
    for (int k = lane; k < K_; k += 32) z += expf(lrow[k] - m);
    for (int off = 16; off > 0; off >>= 1) z += __shfl_xor(z, off);
    if (lane == 0) p[row] = expf(lrow[maxid[b]] - m) / z;
}

// ---------------------------------------------------------------------------
// 6) Per-batch full bitonic argsort of p (descending, stable on ties).
//    64-bit key: hi = ~orderable(p) (descending p), lo = index (ascending).
// ---------------------------------------------------------------------------
__global__ void __launch_bounds__(1024)
sort_kernel(const float* __restrict__ p, int* __restrict__ ranks)
{
    __shared__ unsigned long long key[S_];       // 32 KB LDS
    int b = blockIdx.x, tid = threadIdx.x;
    for (int i = tid; i < S_; i += 1024) {
        unsigned u = __float_as_uint(p[b * S_ + i]);
        unsigned o = (u & 0x80000000u) ? ~u : (u | 0x80000000u);
        key[i] = ((unsigned long long)(~o) << 32) | (unsigned)i;
    }
    __syncthreads();
    for (int k = 2; k <= S_; k <<= 1) {
        for (int j = k >> 1; j > 0; j >>= 1) {
            for (int i = tid; i < S_; i += 1024) {
                int ixj = i ^ j;
                if (ixj > i) {
                    bool up = ((i & k) == 0);
                    unsigned long long a = key[i], c = key[ixj];
                    if ((a > c) == up) { key[i] = c; key[ixj] = a; }
                }
            }
            __syncthreads();
        }
    }
    for (int i = tid; i < S_; i += 1024)
        ranks[b * S_ + i] = (int)(key[i] & 0xffffffffu);
}

// 7) Gathers + weights/mask
__global__ void gather_selections(const float* __restrict__ x, const int* __restrict__ ranks,
                                  float* __restrict__ sel)
{
    int i = blockIdx.x, b = blockIdx.y, t = threadIdx.x;      // 128 threads
    int s = ranks[b * S_ + i];
    const float4* src = (const float4*)(x + ((size_t)b * S_ + s) * C_);
    float4* dst = (float4*)(sel + ((size_t)b * NSEL + i) * C_);
    dst[t] = src[t];
}

__global__ void gather_preds(const float* __restrict__ logits, const int* __restrict__ ranks,
                             float* __restrict__ preds1, float* __restrict__ preds0)
{
    int r = blockIdx.x, b = blockIdx.y, t = threadIdx.x;      // 64 threads
    int s = ranks[b * S_ + r];
    const float* src = logits + ((size_t)b * S_ + s) * KP;
    float* dst = (r < NSEL) ? (preds1 + ((size_t)b * NSEL + r) * K_)
                            : (preds0 + ((size_t)b * (S_ - NSEL) + (r - NSEL)) * K_);
    for (int k = t; k < K_; k += 64) dst[k] = src[k];
}

__global__ void weights_mask(const float* __restrict__ p, const int* __restrict__ ranks,
                             float* __restrict__ weights, float* __restrict__ mask)
{
    int i = blockIdx.x * 256 + threadIdx.x;       // i = b*S_ + rankpos
    int b = i >> 12, r = i & (S_ - 1);
    int s    = ranks[b * S_ + r];
    int sthr = ranks[b * S_ + (NSEL - 1)];
    float thr = p[b * S_ + sthr];
    float pv  = p[b * S_ + s];
    mask[(size_t)b * S_ + s]    = (r < NSEL) ? 1.0f : 0.0f;
    weights[(size_t)b * S_ + s] = (pv >= thr) ? 1.0f : 0.0f;
}

// ---------------------------------------------------------------------------
extern "C" void kernel_launch(void* const* d_in, const int* in_sizes, int n_in,
                              void* d_out, int out_size, void* d_ws, size_t ws_size,
                              hipStream_t stream)
{
    const float* x    = (const float*)d_in[0];
    const float* W    = (const float*)d_in[1];
    const float* bias = (const float*)d_in[2];

    char* ws = (char*)d_ws;
    float*  logits = (float*)(ws + OFF_LOGITS);
    __bf16* wfrag  = (__bf16*)(ws + OFF_WFRAG);
    float*  colsum = (float*)(ws + OFF_COLSUM);
    int*    maxid  = (int*)(ws + OFF_MAXID);
    float*  p      = (float*)(ws + OFF_P);
    int*    ranks  = (int*)(ws + OFF_RANKS);

    float* out     = (float*)d_out;
    float* sel     = out;                                   // [32,256,512]
    float* preds1  = sel    + (size_t)B_ * NSEL * C_;       // [32,256,200]
    float* preds0  = preds1 + (size_t)B_ * NSEL * K_;       // [32,3840,200]
    float* weights = preds0 + (size_t)B_ * (S_ - NSEL) * K_;// [32,4096]
    float* mask    = weights + (size_t)B_ * S_;             // [32,4096]

    prep_wfrag      <<<dim3(NKT, NCT), 32, 0, stream>>>(W, wfrag);
    zero_colsum     <<<B_, 256, 0, stream>>>(colsum);
    gemm_wmma       <<<NROWS / 16 / 8, 256, 0, stream>>>(x, bias, wfrag, logits);
    colsum_kernel   <<<dim3(B_, 16), 256, 0, stream>>>(logits, colsum);
    argmax_kernel   <<<B_, 32, 0, stream>>>(colsum, maxid);
    row_softmax     <<<NROWS / 8, 256, 0, stream>>>(logits, maxid, p);
    sort_kernel     <<<B_, 1024, 0, stream>>>(p, ranks);
    gather_selections<<<dim3(NSEL, B_), 128, 0, stream>>>(x, ranks, sel);
    gather_preds    <<<dim3(S_, B_), 64, 0, stream>>>(logits, ranks, preds1, preds0);
    weights_mask    <<<NROWS / 256, 256, 0, stream>>>(p, ranks, weights, mask);
}
